// BinaryXnorExceptOutliersLinearActivationColumn_30416958390731
// MI455X (gfx1250) — matrix operand order repaired and checked by reading, not compiled, not run on hardware
//
#include <hip/hip_runtime.h>

// ---------------------------------------------------------------------------
// out = x @ w_bin^T + bias, w_bin = (mask ? w*scale_row : w).
// Compute-bound GEMM (~1.7 kFLOP/byte) -> bf16 WMMA pipe (16x16x32) with a
// bf16x3 split (hi*hi + hi*lo + lo*hi, f32 accum) for ~fp32 accuracy.
// Preconverted path copies tiles global->LDS with GLOBAL_LOAD_ASYNC_TO_LDS
// (ASYNCcnt DMA, zero VGPR staging, overlaps WMMA); fallback is a spill-free
// register-staged copy phase after the WMMA phase.
// ---------------------------------------------------------------------------

#if defined(__has_builtin)
#if __has_builtin(__builtin_amdgcn_global_load_async_to_lds_b128) && \
    __has_builtin(__builtin_amdgcn_s_wait_asynccnt)
#define USE_ASYNC_COPY 1
#endif
#endif
#ifndef USE_ASYNC_COPY
#define USE_ASYNC_COPY 0
#endif

typedef __attribute__((ext_vector_type(16))) __bf16 v16bf;
typedef __attribute__((ext_vector_type(8)))  float  v8f;

#define TILE_M 128
#define TILE_N 128
#define TILE_K 32
#define LDST   40                          // padded LDS row stride (bf16 elems, 80B)
#define SA_ELEMS (2 * 2 * TILE_M * LDST)   // [buf][hi/lo][row][col]
#define SB_ELEMS (2 * 2 * TILE_N * LDST)

union FragBF {
  uint4 q[2];
  v16bf v;
};

#if USE_ASYNC_COPY
// Exact pointee type from the builtin's prototype (per compiler diagnostic):
// pointer to 'int __attribute__((__vector_size__(4*sizeof(int))))'.
typedef int v4i_pay __attribute__((__vector_size__(4 * sizeof(int))));
typedef __attribute__((address_space(1))) v4i_pay as1_v4i;  // global ("__device__")
typedef __attribute__((address_space(3))) v4i_pay as3_v4i;  // LDS ("__shared__")
// Per-lane 16B async DMA: global -> LDS, tracked by ASYNCcnt, no VGPR data.
__device__ __forceinline__ void async_copy_b128(const void* gsrc, void* ldst) {
  as1_v4i* g = (as1_v4i*)(unsigned long long)(__SIZE_TYPE__)gsrc;
  // generic LDS address: high bits = shared aperture, low 32 = LDS offset
  as3_v4i* l = (as3_v4i*)(unsigned)(__SIZE_TYPE__)ldst;
  __builtin_amdgcn_global_load_async_to_lds_b128(g, l, 0, 0);
}
#endif

// f32 pair -> packed bf16 hi word + residual lo word (hardware cvt, RNE)
__device__ __forceinline__ void split2(float a, float b, unsigned &hi, unsigned &lo) {
  __bf16 ha = (__bf16)a, hb = (__bf16)b;
  float  ra = a - (float)ha, rb = b - (float)hb;
  __bf16 la = (__bf16)ra, lb = (__bf16)rb;
  hi = (unsigned)__builtin_bit_cast(unsigned short, ha) |
       ((unsigned)__builtin_bit_cast(unsigned short, hb) << 16);
  lo = (unsigned)__builtin_bit_cast(unsigned short, la) |
       ((unsigned)__builtin_bit_cast(unsigned short, lb) << 16);
}

// ---------------------------------------------------------------------------
// Preprocessing
// ---------------------------------------------------------------------------
__global__ void col_norm_kernel(const float* __restrict__ w, float* __restrict__ colnorm,
                                int OUT, int IN) {
  int col = blockIdx.x * blockDim.x + threadIdx.x;
  if (col >= IN) return;
  float s = 0.0f;
  for (int r = 0; r < OUT; ++r) s += fabsf(w[(size_t)r * IN + col]);
  colnorm[col] = s;
}

__global__ void quantile_mask_kernel(const float* __restrict__ colnorm,
                                     float* __restrict__ maskf, float* __restrict__ meta,
                                     int IN) {
  __shared__ float sv[4096];
  __shared__ float red[1024];
  __shared__ float s_lo, s_hi;
  const int tid = threadIdx.x;

  for (int i = tid; i < 4096; i += 1024) sv[i] = (i < IN) ? colnorm[i] : 3.4e38f;
  __syncthreads();
  for (int k = 2; k <= 4096; k <<= 1) {
    for (int j = k >> 1; j > 0; j >>= 1) {
      for (int t = tid; t < 4096; t += 1024) {
        int ixj = t ^ j;
        if (ixj > t) {
          bool up = ((t & k) == 0);
          float a = sv[t], b = sv[ixj];
          if ((a > b) == up) { sv[t] = b; sv[ixj] = a; }
        }
      }
      __syncthreads();
    }
  }
  if (tid == 0) {
    float pos = 0.05f * (float)(IN - 1);
    int   i0  = (int)pos;
    float fr  = pos - (float)i0;
    s_lo = sv[i0] + fr * (sv[i0 + 1] - sv[i0]);
    pos = 0.95f * (float)(IN - 1);
    i0  = (int)pos;
    fr  = pos - (float)i0;
    s_hi = sv[i0] + fr * (sv[i0 + 1] - sv[i0]);
    meta[0] = s_lo;
    meta[1] = s_hi;
  }
  __syncthreads();

  float local = 0.0f;
  for (int i = tid; i < IN; i += 1024) {
    float cn  = colnorm[i];
    float b   = ((cn > s_lo) && (cn < s_hi)) ? 0.0f : 1.0f;  // 1 = binarized
    maskf[i]  = b;
    local    += b;
  }
  red[tid] = local;
  __syncthreads();
  for (int s = 512; s > 0; s >>= 1) {
    if (tid < s) red[tid] += red[tid + s];
    __syncthreads();
  }
  if (tid == 0) meta[2] = red[0];
}

__global__ void scale_kernel(const float* __restrict__ w, const float* __restrict__ maskf,
                             const float* __restrict__ meta, float* __restrict__ scale,
                             int IN) {
  __shared__ float red[256];
  const int row = blockIdx.x;
  const int tid = threadIdx.x;
  const float* wr = w + (size_t)row * IN;
  float s = 0.0f;
  for (int i = tid; i < IN; i += 256) s += fabsf(wr[i]) * maskf[i];
  red[tid] = s;
  __syncthreads();
  for (int k = 128; k > 0; k >>= 1) {
    if (tid < k) red[tid] += red[tid + k];
    __syncthreads();
  }
  if (tid == 0) scale[row] = red[0] / meta[2];
}

// ---------------------------------------------------------------------------
// One-time bf16 hi/lo plane conversion (preconverted path)
// ---------------------------------------------------------------------------
__global__ void convert_x_kernel(const float* __restrict__ x,
                                 unsigned short* __restrict__ xh,
                                 unsigned short* __restrict__ xl, long long total4) {
  long long i = (long long)blockIdx.x * blockDim.x + threadIdx.x;  // float4 units
  if (i >= total4) return;
  float4 v = ((const float4*)x)[i];
  unsigned h0, l0, h1, l1;
  split2(v.x, v.y, h0, l0);
  split2(v.z, v.w, h1, l1);
  ((uint2*)xh)[i] = make_uint2(h0, h1);
  ((uint2*)xl)[i] = make_uint2(l0, l1);
}

__global__ void convert_w_kernel(const float* __restrict__ w,
                                 const float* __restrict__ maskf,
                                 const float* __restrict__ scale,
                                 unsigned short* __restrict__ wh,
                                 unsigned short* __restrict__ wl,
                                 int K, long long total4) {
  long long i = (long long)blockIdx.x * blockDim.x + threadIdx.x;  // float4 units
  if (i >= total4) return;
  long long e  = i * 4;
  int       n  = (int)(e / K);
  int       k  = (int)(e % K);
  float4 v  = ((const float4*)w)[i];
  float4 m  = *(const float4*)(maskf + k);
  float  s1 = scale[n] - 1.0f;   // factor = 1 + mask*(scale-1)
  v.x *= fmaf(m.x, s1, 1.0f);
  v.y *= fmaf(m.y, s1, 1.0f);
  v.z *= fmaf(m.z, s1, 1.0f);
  v.w *= fmaf(m.w, s1, 1.0f);
  unsigned h0, l0, h1, l1;
  split2(v.x, v.y, h0, l0);
  split2(v.z, v.w, h1, l1);
  ((uint2*)wh)[i] = make_uint2(h0, h1);
  ((uint2*)wl)[i] = make_uint2(l0, l1);
}

// ---------------------------------------------------------------------------
// WMMA compute: A frags held, B frags streamed in tn-pairs.
// Per pair: 12 WMMAs, same-accumulator reuse distance 4.
// Live set: acc(64) + a(32) + b(32) + addressing -> ~145 VGPRs.
// ---------------------------------------------------------------------------
__device__ __forceinline__ void wmma_step(const unsigned short* sA, const unsigned short* sB,
                                          int buf, int wm, int wn, int half, int l16,
                                          v8f acc[2][4]) {
  FragBF aH[2], aL[2];
#pragma unroll
  for (int tm = 0; tm < 2; ++tm) {
    // A 16x32 bf16: lane-half 0 holds K 0..7 (V0-3) and 16..23 (V4-7)
    const unsigned short* bh = &sA[((size_t)buf * 2 + 0) * TILE_M * LDST +
                                   (size_t)(wm * 32 + tm * 16 + l16) * LDST];
    const unsigned short* bl = &sA[((size_t)buf * 2 + 1) * TILE_M * LDST +
                                   (size_t)(wm * 32 + tm * 16 + l16) * LDST];
    aH[tm].q[0] = *(const uint4*)(bh + half * 8);
    aH[tm].q[1] = *(const uint4*)(bh + 16 + half * 8);
    aL[tm].q[0] = *(const uint4*)(bl + half * 8);
    aL[tm].q[1] = *(const uint4*)(bl + 16 + half * 8);
  }
#pragma unroll
  for (int pr = 0; pr < 2; ++pr) {
    FragBF bH[2], bL[2];
#pragma unroll
    for (int j = 0; j < 2; ++j) {
      const int tn = pr * 2 + j;
      // B 32x16 bf16: lane-half 0 holds K 0..7 (V0-3) and 16..23 (V4-7)
      const unsigned short* bh = &sB[((size_t)buf * 2 + 0) * TILE_N * LDST +
                                     (size_t)(wn * 64 + tn * 16 + l16) * LDST];
      const unsigned short* bl = &sB[((size_t)buf * 2 + 1) * TILE_N * LDST +
                                     (size_t)(wn * 64 + tn * 16 + l16) * LDST];
      bH[j].q[0] = *(const uint4*)(bh + half * 8);
      bH[j].q[1] = *(const uint4*)(bh + 16 + half * 8);
      bL[j].q[0] = *(const uint4*)(bl + half * 8);
      bL[j].q[1] = *(const uint4*)(bl + 16 + half * 8);
    }
#pragma unroll
    for (int j = 0; j < 2; ++j)
#pragma unroll
      for (int tm = 0; tm < 2; ++tm)
        acc[tm][pr * 2 + j] = __builtin_amdgcn_wmma_f32_16x16x32_bf16(
            false, aH[tm].v, false, bH[j].v, (short)0, acc[tm][pr * 2 + j], false, false);
#pragma unroll
    for (int j = 0; j < 2; ++j)
#pragma unroll
      for (int tm = 0; tm < 2; ++tm)
        acc[tm][pr * 2 + j] = __builtin_amdgcn_wmma_f32_16x16x32_bf16(
            false, aH[tm].v, false, bL[j].v, (short)0, acc[tm][pr * 2 + j], false, false);
#pragma unroll
    for (int j = 0; j < 2; ++j)
#pragma unroll
      for (int tm = 0; tm < 2; ++tm)
        acc[tm][pr * 2 + j] = __builtin_amdgcn_wmma_f32_16x16x32_bf16(
            false, aL[tm].v, false, bH[j].v, (short)0, acc[tm][pr * 2 + j], false, false);
  }
}

__device__ __forceinline__ void epilogue(float* __restrict__ out, const float* __restrict__ bias,
                                         v8f acc[2][4], int m0, int n0, int wm, int wn,
                                         int half, int l16, int N) {
#pragma unroll
  for (int tm = 0; tm < 2; ++tm)
#pragma unroll
    for (int tn = 0; tn < 4; ++tn) {
      const int mg = m0 + wm * 32 + tm * 16 + half * 8;  // C: M = r + 8*(lane/16)
      const int ng = n0 + wn * 64 + tn * 16 + l16;       //    N = lane%16
      const float bv = bias[ng];
#pragma unroll
      for (int r = 0; r < 8; ++r)
        out[(size_t)(mg + r) * N + ng] = acc[tm][tn][r] + bv;
    }
}

// ---------------------------------------------------------------------------
// GEMM path A: preconverted bf16 planes.
// ---------------------------------------------------------------------------
__global__ __launch_bounds__(256)
__attribute__((amdgpu_waves_per_eu(1)))
void gemm_pre_bf16x3(const unsigned short* __restrict__ xh, const unsigned short* __restrict__ xl,
                     const unsigned short* __restrict__ wh, const unsigned short* __restrict__ wl,
                     const float* __restrict__ bias, float* __restrict__ out,
                     int M, int N, int K) {
  extern __shared__ __align__(16) unsigned short smem[];
  unsigned short* sA = smem;
  unsigned short* sB = smem + SA_ELEMS;

  const int tid  = threadIdx.x;
  const int lane = tid & 31;
  const int wave = tid >> 5;
  const int wm   = wave & 3;
  const int wn   = wave >> 2;
  const int half = lane >> 4;
  const int l16  = lane & 15;
  const int m0   = blockIdx.y * TILE_M;
  const int n0   = blockIdx.x * TILE_N;

  const int lrow   = tid >> 2;        // 0..63 (row within 64-row pass)
  const int lchunk = (tid & 3) << 3;  // 0,8,16,24 (ushort offset, 16B chunks)

  v8f zero = {0.f, 0.f, 0.f, 0.f, 0.f, 0.f, 0.f, 0.f};
  v8f acc[2][4];
#pragma unroll
  for (int i = 0; i < 2; ++i)
#pragma unroll
    for (int j = 0; j < 4; ++j) acc[i][j] = zero;

#if USE_ASYNC_COPY
  // DMA copy: 8 async b128 per thread, no VGPR staging; overlaps with WMMA.
  auto copy_tile = [&](int s, int buf) {
    const int kw = s * TILE_K;
#pragma unroll
    for (int p = 0; p < 2; ++p) {
      const int r = p * 64 + lrow;
      const size_t gx = (size_t)(m0 + r) * K + kw + lchunk;
      const size_t gw = (size_t)(n0 + r) * K + kw + lchunk;
      async_copy_b128(xh + gx, &sA[((size_t)buf * 2 + 0) * TILE_M * LDST + (size_t)r * LDST + lchunk]);
      async_copy_b128(xl + gx, &sA[((size_t)buf * 2 + 1) * TILE_M * LDST + (size_t)r * LDST + lchunk]);
      async_copy_b128(wh + gw, &sB[((size_t)buf * 2 + 0) * TILE_N * LDST + (size_t)r * LDST + lchunk]);
      async_copy_b128(wl + gw, &sB[((size_t)buf * 2 + 1) * TILE_N * LDST + (size_t)r * LDST + lchunk]);
    }
  };

  const int kSteps = K / TILE_K;
  copy_tile(0, 0);
  __builtin_amdgcn_s_wait_asynccnt(0);
  __syncthreads();
  for (int s = 0; s < kSteps; ++s) {
    if (s + 1 < kSteps) copy_tile(s + 1, (s + 1) & 1);  // DMA in flight during WMMA
    wmma_step(sA, sB, s & 1, wm, wn, half, l16, acc);
    if (s + 1 < kSteps) __builtin_amdgcn_s_wait_asynccnt(0);
    __syncthreads();
  }
#else
  // Register-staged copy, placed after WMMA phase (no spill pressure overlap).
  auto copy_tile = [&](int s, int buf) {
    const int kw = s * TILE_K;
    uint4 pah[2], pal[2], pbh[2], pbl[2];
#pragma unroll
    for (int p = 0; p < 2; ++p) {
      const int r = p * 64 + lrow;
      pah[p] = *(const uint4*)(xh + (size_t)(m0 + r) * K + kw + lchunk);
      pal[p] = *(const uint4*)(xl + (size_t)(m0 + r) * K + kw + lchunk);
      pbh[p] = *(const uint4*)(wh + (size_t)(n0 + r) * K + kw + lchunk);
      pbl[p] = *(const uint4*)(wl + (size_t)(n0 + r) * K + kw + lchunk);
    }
#pragma unroll
    for (int p = 0; p < 2; ++p) {
      const int r = p * 64 + lrow;
      *(uint4*)&sA[((size_t)buf * 2 + 0) * TILE_M * LDST + (size_t)r * LDST + lchunk] = pah[p];
      *(uint4*)&sA[((size_t)buf * 2 + 1) * TILE_M * LDST + (size_t)r * LDST + lchunk] = pal[p];
      *(uint4*)&sB[((size_t)buf * 2 + 0) * TILE_N * LDST + (size_t)r * LDST + lchunk] = pbh[p];
      *(uint4*)&sB[((size_t)buf * 2 + 1) * TILE_N * LDST + (size_t)r * LDST + lchunk] = pbl[p];
    }
  };

  const int kSteps = K / TILE_K;
  copy_tile(0, 0);
  __syncthreads();
  for (int s = 0; s < kSteps; ++s) {
    wmma_step(sA, sB, s & 1, wm, wn, half, l16, acc);
    if (s + 1 < kSteps) copy_tile(s + 1, (s + 1) & 1);
    __syncthreads();
  }
#endif
  epilogue(out, bias, acc, m0, n0, wm, wn, half, l16, N);
}

// ---------------------------------------------------------------------------
// GEMM path B (fallback, small ws): fused on-the-fly split + mask/scale.
// (Needs register-space transform, so no async DMA here.)
// ---------------------------------------------------------------------------
__global__ __launch_bounds__(256)
__attribute__((amdgpu_waves_per_eu(1)))
void gemm_fused_bf16x3(const float* __restrict__ x, const float* __restrict__ w,
                       const float* __restrict__ bias, const float* __restrict__ maskf,
                       const float* __restrict__ scale, float* __restrict__ out,
                       int M, int N, int K) {
  extern __shared__ __align__(16) unsigned short smem[];
  unsigned short* sA = smem;
  unsigned short* sB = smem + SA_ELEMS;

  const int tid  = threadIdx.x;
  const int lane = tid & 31;
  const int wave = tid >> 5;
  const int wm   = wave & 3;
  const int wn   = wave >> 2;
  const int half = lane >> 4;
  const int l16  = lane & 15;
  const int m0   = blockIdx.y * TILE_M;
  const int n0   = blockIdx.x * TILE_N;

  const int lrow = tid >> 3;        // 0..31
  const int lcol = (tid & 7) << 2;  // 0..28 step 4

  float psc1[4];
#pragma unroll
  for (int p = 0; p < 4; ++p) psc1[p] = scale[n0 + p * 32 + lrow] - 1.0f;

  v8f zero = {0.f, 0.f, 0.f, 0.f, 0.f, 0.f, 0.f, 0.f};
  v8f acc[2][4];
#pragma unroll
  for (int i = 0; i < 2; ++i)
#pragma unroll
    for (int j = 0; j < 4; ++j) acc[i][j] = zero;

  auto copy_tile = [&](int s, int buf) {
    const int kw = s * TILE_K;
    float4 pa[4], pb[4];
    float4 pmask = *(const float4*)(maskf + kw + lcol);
#pragma unroll
    for (int p = 0; p < 4; ++p) {
      const int r = p * 32 + lrow;
      pa[p] = *(const float4*)(x + (size_t)(m0 + r) * K + kw + lcol);
      pb[p] = *(const float4*)(w + (size_t)(n0 + r) * K + kw + lcol);
    }
#pragma unroll
    for (int p = 0; p < 4; ++p) {
      const int r = p * 32 + lrow;
      unsigned h0, l0, h1, l1;
      split2(pa[p].x, pa[p].y, h0, l0);
      split2(pa[p].z, pa[p].w, h1, l1);
      *(uint2*)&sA[((size_t)buf * 2 + 0) * TILE_M * LDST + (size_t)r * LDST + lcol] = make_uint2(h0, h1);
      *(uint2*)&sA[((size_t)buf * 2 + 1) * TILE_M * LDST + (size_t)r * LDST + lcol] = make_uint2(l0, l1);
      const float s1 = psc1[p];
      float4 v = pb[p];
      v.x *= fmaf(pmask.x, s1, 1.0f);
      v.y *= fmaf(pmask.y, s1, 1.0f);
      v.z *= fmaf(pmask.z, s1, 1.0f);
      v.w *= fmaf(pmask.w, s1, 1.0f);
      split2(v.x, v.y, h0, l0);
      split2(v.z, v.w, h1, l1);
      *(uint2*)&sB[((size_t)buf * 2 + 0) * TILE_N * LDST + (size_t)r * LDST + lcol] = make_uint2(h0, h1);
      *(uint2*)&sB[((size_t)buf * 2 + 1) * TILE_N * LDST + (size_t)r * LDST + lcol] = make_uint2(l0, l1);
    }
  };

  const int kSteps = K / TILE_K;
  copy_tile(0, 0);
  __syncthreads();
  for (int s = 0; s < kSteps; ++s) {
    wmma_step(sA, sB, s & 1, wm, wn, half, l16, acc);
    if (s + 1 < kSteps) copy_tile(s + 1, (s + 1) & 1);
    __syncthreads();
  }
  epilogue(out, bias, acc, m0, n0, wm, wn, half, l16, N);
}

// ---------------------------------------------------------------------------
extern "C" void kernel_launch(void* const* d_in, const int* in_sizes, int n_in,
                              void* d_out, int out_size, void* d_ws, size_t ws_size,
                              hipStream_t stream) {
  const float* x    = (const float*)d_in[0];
  const float* w    = (const float*)d_in[1];
  const float* bias = (const float*)d_in[2];
  float*       out  = (float*)d_out;

  const int OUT = in_sizes[2];        // out features (4096)
  const int IN  = in_sizes[1] / OUT;  // in features (4096)
  const int M   = in_sizes[0] / IN;   // B*S (8192)
  const int N   = OUT;
  const int K   = IN;

  float* ws      = (float*)d_ws;
  float* colnorm = ws;                  // [IN]
  float* maskf   = ws + IN;             // [IN]
  float* scale   = ws + 2 * IN;         // [OUT]
  float* meta    = ws + 2 * IN + OUT;   // [4]

  col_norm_kernel<<<(IN + 255) / 256, 256, 0, stream>>>(w, colnorm, OUT, IN);
  quantile_mask_kernel<<<1, 1024, 0, stream>>>(colnorm, maskf, meta, IN);
  scale_kernel<<<OUT, 256, 0, stream>>>(w, maskf, meta, scale, IN);

  const size_t small_bytes = (size_t)(2 * IN + OUT + 8) * sizeof(float);
  const size_t plane_off   = (small_bytes + 255) & ~(size_t)255;
  const size_t planeX      = (size_t)M * K * sizeof(unsigned short);
  const size_t planeW      = (size_t)N * K * sizeof(unsigned short);
  const size_t need        = plane_off + 2 * planeX + 2 * planeW;

  dim3 grid(N / TILE_N, M / TILE_M);
  const size_t lds_bytes = (size_t)(SA_ELEMS + SB_ELEMS) * sizeof(unsigned short);  // 80 KB

  if (ws_size >= need) {
    unsigned short* xh = (unsigned short*)((char*)d_ws + plane_off);
    unsigned short* xl = xh + (size_t)M * K;
    unsigned short* wh = xl + (size_t)M * K;
    unsigned short* wl = wh + (size_t)N * K;
    const long long x4 = (long long)M * K / 4;
    const long long w4 = (long long)N * K / 4;
    convert_x_kernel<<<(unsigned)((x4 + 255) / 256), 256, 0, stream>>>(x, xh, xl, x4);
    convert_w_kernel<<<(unsigned)((w4 + 255) / 256), 256, 0, stream>>>(w, maskf, scale, wh, wl, K, w4);
    gemm_pre_bf16x3<<<grid, 256, lds_bytes, stream>>>(xh, xl, wh, wl, bias, out, M, N, K);
  } else {
    gemm_fused_bf16x3<<<grid, 256, lds_bytes, stream>>>(x, w, bias, maskf, scale, out, M, N, K);
  }
}
